// HeadAttention_8839042695282
// MI455X (gfx1250) — matrix operand order repaired and checked by reading, not compiled, not run on hardware
//
#include <hip/hip_runtime.h>
#include <hip/hip_bf16.h>

typedef __bf16 bf16;
typedef __attribute__((ext_vector_type(16))) __bf16 v16bf;
typedef __attribute__((ext_vector_type(8)))  __bf16 v8bf;
typedef __attribute__((ext_vector_type(8)))  float  v8f;
typedef __attribute__((ext_vector_type(4)))  int    i32x4;

#define EMB     512
#define DH      512
#define SEQ     2048
#define BATCH   4
#define KV_TILE 32
#define VPAD    520   // padded LDS row stride (bf16 elems) to break bank conflicts

// ---- async global->LDS path (gfx1250 GLOBAL_LOAD_ASYNC_TO_LDS_B128, ASYNCcnt) ----
#if defined(__has_builtin)
# if __has_builtin(__builtin_amdgcn_global_load_async_to_lds_b128) && \
     __has_builtin(__builtin_amdgcn_s_wait_asynccnt)
#  define HAVE_ASYNC_LDS 1
# endif
#endif
#ifndef HAVE_ASYNC_LDS
# define HAVE_ASYNC_LDS 0
#endif

#if HAVE_ASYNC_LDS
typedef __attribute__((address_space(1))) i32x4 gi32x4;  // global v4i32
typedef __attribute__((address_space(3))) i32x4 li32x4;  // LDS v4i32
#endif

// ---- fragment builders (CDNA5 wave32 WMMA layouts, 16-bit A/B) ----
// A 16x32: lane L holds row M=L%16; VGPR0-3 = K in [base, base+8), VGPR4-7 = K in
// [base+8, base+16), base = (L>=16 ? 16 : 0) (contiguous k order). Chunks are contiguous in memory.
__device__ __forceinline__ v16bf ld_cvt_pair8(const float* __restrict__ p0,
                                              const float* __restrict__ p1) {
  float4 a0 = ((const float4*)p0)[0];
  float4 a1 = ((const float4*)p0)[1];
  float4 b0 = ((const float4*)p1)[0];
  float4 b1 = ((const float4*)p1)[1];
  v16bf r;
  r[0]=(bf16)a0.x; r[1]=(bf16)a0.y; r[2]=(bf16)a0.z;  r[3]=(bf16)a0.w;
  r[4]=(bf16)a1.x; r[5]=(bf16)a1.y; r[6]=(bf16)a1.z;  r[7]=(bf16)a1.w;
  r[8]=(bf16)b0.x; r[9]=(bf16)b0.y; r[10]=(bf16)b0.z; r[11]=(bf16)b0.w;
  r[12]=(bf16)b1.x;r[13]=(bf16)b1.y;r[14]=(bf16)b1.z; r[15]=(bf16)b1.w;
  return r;
}
// B 32x16: lane L holds column N=L%16; lanes 0-15 K=0..15, lanes 16-31 K=16..31.
__device__ __forceinline__ v16bf ld_cvt_16(const float* __restrict__ p) {
  return ld_cvt_pair8(p, p + 8);
}
__device__ __forceinline__ v16bf ld_bf_pair8(const bf16* __restrict__ p0,
                                             const bf16* __restrict__ p1) {
  v8bf a = *(const v8bf*)p0;
  v8bf b = *(const v8bf*)p1;
  v16bf r;
#pragma unroll
  for (int i = 0; i < 8; ++i) { r[i] = a[i]; r[i + 8] = b[i]; }
  return r;
}

// ---------------- Kernel 1: fused QKV projection (y = x @ W^T + b) ----------------
// grid: (M/128, D/64, 3), block: 256 (8 waves). Wave -> 16x64 tile, K-loop over EMB.
__global__ __launch_bounds__(256) void qkv_proj(
    const float* __restrict__ x,
    const float* __restrict__ Wq, const float* __restrict__ bq,
    const float* __restrict__ Wk, const float* __restrict__ bk,
    const float* __restrict__ Wv, const float* __restrict__ bv,
    bf16* __restrict__ Q, bf16* __restrict__ K, bf16* __restrict__ V) {
  const int wave = threadIdx.x >> 5, lane = threadIdx.x & 31;
  const int half = lane >> 4, lr = lane & 15;
  const int m0 = blockIdx.x * 128 + wave * 16;
  const int n0 = blockIdx.y * 64;
  const int z  = blockIdx.z;
  const float* W    = (z == 0) ? Wq : (z == 1) ? Wk : Wv;
  const float* bias = (z == 0) ? bq : (z == 1) ? bk : bv;
  bf16* out         = (z == 0) ? Q  : (z == 1) ? K  : V;

  const size_t arow = (size_t)(m0 + lr) * EMB;
  v8f acc[4] = {};
  for (int k0 = 0; k0 < EMB; k0 += 32) {
    v16bf af = ld_cvt_pair8(&x[arow + k0 + half * 16],
                            &x[arow + k0 + half * 16 + 8]);
#pragma unroll
    for (int nt = 0; nt < 4; ++nt) {
      const int n = n0 + nt * 16 + lr;
      v16bf bfr = ld_cvt_16(&W[(size_t)n * EMB + k0 + half * 16]);
      acc[nt] = __builtin_amdgcn_wmma_f32_16x16x32_bf16(
          false, af, false, bfr, (short)0, acc[nt], false, false);
    }
  }
#pragma unroll
  for (int nt = 0; nt < 4; ++nt) {
    const int n = n0 + nt * 16 + lr;
    const float b = bias[n];
#pragma unroll
    for (int v = 0; v < 8; ++v) {
      const int m = m0 + v + half * 8;
      out[(size_t)m * DH + n] = (bf16)(acc[nt][v] + b);
    }
  }
}

// ---------------- Kernel 2: causal flash attention ----------------
// grid: (S/16, B), block 256 (8 waves). Block owns 16 query rows; wave w owns
// head-dim slice [64w, 64w+64). V key-tile staged to LDS via async copies.
__global__ __launch_bounds__(256) void attn(
    const bf16* __restrict__ Q, const bf16* __restrict__ K,
    const bf16* __restrict__ V, float* __restrict__ out) {
  __shared__ __align__(16) bf16 vtile[KV_TILE][VPAD];  // 32.5 KB
  __shared__ float part[8][16][KV_TILE];               // 16 KB
  __shared__ float sc[16][KV_TILE];                    // 2 KB
  __shared__ float mrow[16], lrow[16], arw[16];

  const int tid = threadIdx.x;
  const int wave = tid >> 5, lane = tid & 31;
  const int half = lane >> 4, lr = lane & 15;
  const int q0 = blockIdx.x * 16;
  const int b  = blockIdx.y;
  const int dw = wave * 64;

  const bf16* Qb = Q + (size_t)b * SEQ * DH;
  const bf16* Kb = K + (size_t)b * SEQ * DH;
  const bf16* Vb = V + (size_t)b * SEQ * DH;

  if (tid < 16) { mrow[tid] = -3.0e38f; lrow[tid] = 0.0f; }

  // Preload Q fragments for this wave's d-slice (constant over the kv loop)
  v16bf qf[2];
#pragma unroll
  for (int kk = 0; kk < 2; ++kk) {
    const bf16* p = &Qb[(size_t)(q0 + lr) * DH + dw + kk * 32 + half * 16];
    qf[kk] = ld_bf_pair8(p, p + 8);
  }

  v8f oacc[4] = {};
  const float scale = 0.044194173824159216f;  // 1/sqrt(512)
  __syncthreads();

  for (int kv0 = 0; kv0 < q0 + 16; kv0 += KV_TILE) {
    // (0) stage V[kv0 .. kv0+31][0..511] into LDS; overlapped with (1)-(3)
    {
      const int r = tid >> 3;  // key row 0..31
      const int s = tid & 7;   // 64-elem segment of the 512-wide row
      const bf16* gp = &Vb[(size_t)(kv0 + r) * DH + s * 64];
      bf16* lp = &vtile[r][s * 64];
#if HAVE_ASYNC_LDS
#pragma unroll
      for (int c = 0; c < 8; ++c)
        __builtin_amdgcn_global_load_async_to_lds_b128(
            (gi32x4*)(const void*)(gp + c * 8),
            (li32x4*)(void*)(lp + c * 8), 0, 0);
#else
#pragma unroll
      for (int c = 0; c < 8; ++c)
        *(v8bf*)(lp + c * 8) = *(const v8bf*)(gp + c * 8);
#endif
    }

    // (1) partial scores S_w = Q[:, d_w] @ K[kv tile, d_w]^T  (2 n-tiles of 16 keys)
    v8f sacc[2] = {};
#pragma unroll
    for (int kk = 0; kk < 2; ++kk) {
#pragma unroll
      for (int nt = 0; nt < 2; ++nt) {
        const bf16* kp =
            &Kb[(size_t)(kv0 + nt * 16 + lr) * DH + dw + kk * 32 + half * 16];
        v16bf kf = *(const v16bf*)kp;  // 32B contiguous: row of K = column of K^T
        sacc[nt] = __builtin_amdgcn_wmma_f32_16x16x32_bf16(
            false, qf[kk], false, kf, (short)0, sacc[nt], false, false);
      }
    }
#pragma unroll
    for (int nt = 0; nt < 2; ++nt)
#pragma unroll
      for (int v = 0; v < 8; ++v)
        part[wave][v + half * 8][nt * 16 + lr] = sacc[nt][v];
    __syncthreads();

    // (2) cross-wave reduce + scale + causal mask (256 thr x 2 elems)
#pragma unroll
    for (int i = 0; i < 2; ++i) {
      const int idx = tid * 2 + i;
      const int r = idx >> 5, c = idx & 31;
      float s = 0.0f;
#pragma unroll
      for (int w = 0; w < 8; ++w) s += part[w][r][c];
      s *= scale;
      if (kv0 + c > q0 + r) s = -3.0e38f;
      sc[r][c] = s;
    }
    __syncthreads();

    // (3) online softmax, one thread per query row
    if (tid < 16) {
      const int r = tid;
      const float mold = mrow[r];
      float mx = mold;
      for (int c = 0; c < KV_TILE; ++c) mx = fmaxf(mx, sc[r][c]);
      const float alpha = __expf(mold - mx);
      float sum = 0.0f;
      for (int c = 0; c < KV_TILE; ++c) {
        const float p = __expf(sc[r][c] - mx);
        sc[r][c] = p;
        sum += p;
      }
      mrow[r] = mx;
      lrow[r] = lrow[r] * alpha + sum;
      arw[r]  = alpha;
    }
    // every wave drains its own async copies BEFORE the barrier, so after the
    // barrier the whole vtile (written by all waves) is visible to everyone
#if HAVE_ASYNC_LDS
    __builtin_amdgcn_s_wait_asynccnt(0);
#endif
    __syncthreads();

    // (4) rescale running O accumulators
#pragma unroll
    for (int v = 0; v < 8; ++v) {
      const float a = arw[v + half * 8];
#pragma unroll
      for (int nt = 0; nt < 4; ++nt) oacc[nt][v] *= a;
    }

    // prefetch next K tile into cache (global_prefetch_b8)
    if (kv0 + KV_TILE < q0 + 16)
      __builtin_prefetch(&Kb[(size_t)(kv0 + KV_TILE + lane) * DH + dw], 0, 1);

    // (5) O_w += P @ V[:, d_w]  (single K=32 step; V columns come from LDS)
    {
      v16bf pf;
#pragma unroll
      for (int j = 0; j < 8; ++j) {
        pf[j]     = (bf16)sc[lr][half * 16 + j];
        pf[j + 8] = (bf16)sc[lr][half * 16 + 8 + j];
      }
#pragma unroll
      for (int nt = 0; nt < 4; ++nt) {
        const int dcol = dw + nt * 16 + lr;
        v16bf vf;
#pragma unroll
        for (int j = 0; j < 16; ++j)
          vf[j] = vtile[half * 16 + j][dcol];  // ds_load column reads
        oacc[nt] = __builtin_amdgcn_wmma_f32_16x16x32_bf16(
            false, pf, false, vf, (short)0, oacc[nt], false, false);
      }
    }
    __syncthreads();  // protect vtile/sc before next iteration overwrites them
  }

  // epilogue: normalize by row sum, store fp32
#pragma unroll
  for (int v = 0; v < 8; ++v) {
    const int m = v + half * 8;
    const float linv = 1.0f / lrow[m];
#pragma unroll
    for (int nt = 0; nt < 4; ++nt)
      out[((size_t)b * SEQ + q0 + m) * DH + dw + nt * 16 + lr] =
          oacc[nt][v] * linv;
  }
}

extern "C" void kernel_launch(void* const* d_in, const int* in_sizes, int n_in,
                              void* d_out, int out_size, void* d_ws, size_t ws_size,
                              hipStream_t stream) {
  const float* x  = (const float*)d_in[0];
  const float* Wq = (const float*)d_in[1];
  const float* bq = (const float*)d_in[2];
  const float* Wk = (const float*)d_in[3];
  const float* bk = (const float*)d_in[4];
  const float* Wv = (const float*)d_in[5];
  const float* bv = (const float*)d_in[6];
  float* out = (float*)d_out;

  const size_t elems = (size_t)BATCH * SEQ * DH;
  bf16* Q = (bf16*)d_ws;
  bf16* K = Q + elems;
  bf16* V = K + elems;

  dim3 g1(BATCH * SEQ / 128, DH / 64, 3);
  qkv_proj<<<g1, 256, 0, stream>>>(x, Wq, bq, Wk, bk, Wv, bv, Q, K, V);

  dim3 g2(SEQ / 16, BATCH);
  attn<<<g2, 256, 0, stream>>>(Q, K, V, out);
}
